// cat3Head_type_49478023250676
// MI455X (gfx1250) — matrix operand order repaired and checked by reading, not compiled, not run on hardware
//
#include <hip/hip_runtime.h>
#include <hip/hip_bf16.h>

typedef __attribute__((ext_vector_type(16))) __bf16 v16bf;
typedef __attribute__((ext_vector_type(8)))  __bf16 v8bf;
typedef __attribute__((ext_vector_type(8)))  float  v8f;

#define NDIM        128     // node feature dim
#define KDIM        384     // 3*NDIM, GEMM K
#define NOUT        768     // 6*NDIM, GEMM N
#define EPB         64      // edges per block = 4 M-tiles of 16
#define MTILES      4       // M-tiles per block, all processed by every wave
#define NTPW        6       // N-tiles per wave (48 / 8 waves), done as 3 pairs
#define LSTRIDE     392     // KDIM + 8 ushort pad (bank stagger); 784B row, 16B aligned
#define FEAT_BYTES  (EPB * LSTRIDE * 2)          // 50176 B
#define LDS_BYTES   (FEAT_BYTES + EPB * 4)       // + 64-float partial array

// ---- fp32 -> bf16 (round-to-nearest-even) ----
__device__ __forceinline__ unsigned short f2bf(float f) {
    unsigned int u = __builtin_bit_cast(unsigned int, f);
    u = (u + 0x7FFFu + ((u >> 16) & 1u)) >> 16;
    return (unsigned short)u;
}
__device__ __forceinline__ unsigned int pk2bf(float a, float b) {
    return (unsigned int)f2bf(a) | ((unsigned int)f2bf(b) << 16);
}

// ---- prepass: W1 fp32 -> bf16 in workspace ----
__global__ __launch_bounds__(256)
void w1_to_bf16_kernel(const float* __restrict__ w1, unsigned short* __restrict__ out, int n) {
    int i = blockIdx.x * 256 + threadIdx.x;
    if (i < n) out[i] = f2bf(w1[i]);
}

// ---- fused edge-MLP: gather -> feat -> WMMA GEMM (A x2-N reuse, B x4-M reuse) ----
__global__ __launch_bounds__(256)
void edge_mlp_wmma_kernel(const float*      __restrict__ x,     // [N_NODES, 128]
                          const long long*  __restrict__ ei,    // [2, E]
                          const unsigned short* __restrict__ w1b, // [768, 384] bf16
                          const float*      __restrict__ b1,    // [768]
                          const float*      __restrict__ w2,    // [768]
                          const float*      __restrict__ b2,    // [1]
                          float*            __restrict__ y,     // [E]
                          int E)
{
    extern __shared__ unsigned char smem[];
    unsigned short* feat     = (unsigned short*)smem;          // [EPB][LSTRIDE] bf16
    float*          spartial = (float*)(smem + FEAT_BYTES);    // [EPB] per-edge partials

    const int tid   = threadIdx.x;
    const int ebase = blockIdx.x * EPB;

    // ---------- Phase 1: build bf16 feature tile in LDS ----------
    {
        const int e  = tid >> 2;          // 0..63: local edge
        const int d0 = (tid & 3) * 32;    // quarter of the 128 dims
        unsigned short* frow = feat + e * LSTRIDE;
        if (tid < EPB) spartial[tid] = 0.0f;
        if (ebase + e < E) {
            const long long s = ei[ebase + e];
            const long long t = ei[E + ebase + e];
            const float4* xs = (const float4*)(x + s * (long long)NDIM);
            const float4* xt = (const float4*)(x + t * (long long)NDIM);
            #pragma unroll
            for (int j = 0; j < 8; ++j) {
                const int d = d0 + j * 4;
                float4 a = xs[d >> 2];
                float4 b = xt[d >> 2];
                // mean
                *(uint2*)(frow + d) = make_uint2(
                    pk2bf((a.x + b.x) * 0.5f, (a.y + b.y) * 0.5f),
                    pk2bf((a.z + b.z) * 0.5f, (a.w + b.w) * 0.5f));
                // product
                *(uint2*)(frow + NDIM + d) = make_uint2(
                    pk2bf(a.x * b.x, a.y * b.y),
                    pk2bf(a.z * b.z, a.w * b.w));
                // squared difference
                float dx = a.x - b.x, dy = a.y - b.y, dz = a.z - b.z, dw = a.w - b.w;
                *(uint2*)(frow + 2 * NDIM + d) = make_uint2(
                    pk2bf(dx * dx, dy * dy),
                    pk2bf(dz * dz, dw * dw));
            }
        }
    }
    __syncthreads();

    // ---------- Phase 2: waves split N (3 pairs of N-tiles), 4 M-tiles each ----------
    const int lane = tid & 31;
    const int wave = tid >> 5;       // 0..7 -> N-tile range [wave*6, wave*6+6)
    const int half = lane >> 4;      // 0 | 1
    const int mr   = lane & 15;      // row (A) / column (B,C) within tile

    const unsigned short* arow[MTILES];
    #pragma unroll
    for (int mt = 0; mt < MTILES; ++mt)
        arow[mt] = feat + (mt * 16 + mr) * LSTRIDE + half * 8;

    float accy[MTILES][8];
    #pragma unroll
    for (int mt = 0; mt < MTILES; ++mt)
        #pragma unroll
        for (int r = 0; r < 8; ++r) accy[mt][r] = 0.0f;

    for (int t = 0; t < NTPW / 2; ++t) {
        const int n0 = (wave * NTPW + t * 2) * 16 + mr;   // first N-tile column
        const int n1 = n0 + 16;                           // second N-tile column
        const unsigned short* bb0 = w1b + n0 * KDIM + half * 8;
        const unsigned short* bb1 = w1b + n1 * KDIM + half * 8;

        v8f c[2][MTILES];
        #pragma unroll
        for (int s = 0; s < 2; ++s)
            #pragma unroll
            for (int mt = 0; mt < MTILES; ++mt) c[s][mt] = (v8f){};

        #pragma unroll 2
        for (int kk = 0; kk < KDIM; kk += 32) {
            // B (32x16): lanes 0-15 hold K {kk..kk+7, kk+16..kk+23}, lanes 16-31 hold K {kk+8..kk+15, kk+24..kk+31}
            union { v16bf v; v8bf h[2]; } B0, B1;
            B0.h[0] = *(const v8bf*)(bb0 + kk);
            B0.h[1] = *(const v8bf*)(bb0 + kk + 16);
            B1.h[0] = *(const v8bf*)(bb1 + kk);
            B1.h[1] = *(const v8bf*)(bb1 + kk + 16);
            // A (16-bit 16x32): lanes 0-15 -> K {kk..kk+7, kk+16..kk+23},
            //                   lanes 16-31 -> K {kk+8..kk+15, kk+24..kk+31}
            union { v16bf v; v8bf h[2]; } A[MTILES];
            #pragma unroll
            for (int mt = 0; mt < MTILES; ++mt) {
                A[mt].h[0] = *(const v8bf*)(arow[mt] + kk);
                A[mt].h[1] = *(const v8bf*)(arow[mt] + kk + 16);
            }
            // 8 WMMAs per A/B fetch: A reused across 2 N-tiles, B across 4 M-tiles
            #pragma unroll
            for (int mt = 0; mt < MTILES; ++mt)
                c[0][mt] = __builtin_amdgcn_wmma_f32_16x16x32_bf16(
                               false, A[mt].v, false, B0.v, (short)0, c[0][mt], false, false);
            #pragma unroll
            for (int mt = 0; mt < MTILES; ++mt)
                c[1][mt] = __builtin_amdgcn_wmma_f32_16x16x32_bf16(
                               false, A[mt].v, false, B1.v, (short)0, c[1][mt], false, false);
        }

        // epilogue: bias + relu + scale by W2[n]; C layout: VGPR r holds
        // row m=r (lanes 0-15) / m=r+8 (lanes 16-31), column n = lane&15
        #pragma unroll
        for (int s = 0; s < 2; ++s) {
            const int n   = (s == 0) ? n0 : n1;
            const float b1n = b1[n];
            const float w2n = w2[n];
            #pragma unroll
            for (int mt = 0; mt < MTILES; ++mt) {
                #pragma unroll
                for (int r = 0; r < 8; ++r) {
                    float h = c[s][mt][r] + b1n;
                    h = h > 0.0f ? h : 0.0f;
                    accy[mt][r] += h * w2n;
                }
            }
        }
    }

    // reduce across the 16 n-lanes of each half (masks < 16 stay in-half)
    #pragma unroll
    for (int m = 1; m < 16; m <<= 1) {
        #pragma unroll
        for (int mt = 0; mt < MTILES; ++mt)
            #pragma unroll
            for (int r = 0; r < 8; ++r)
                accy[mt][r] += __shfl_xor(accy[mt][r], m, 32);
    }
    // lanes mr==0 (lane 0 and lane 16) hold per-edge sums for this wave's N-range;
    // combine across the 8 waves via DS_ADD_F32 in LDS
    if (mr == 0) {
        #pragma unroll
        for (int mt = 0; mt < MTILES; ++mt)
            #pragma unroll
            for (int r = 0; r < 8; ++r)
                atomicAdd(&spartial[mt * 16 + half * 8 + r], accy[mt][r]);
    }
    __syncthreads();

    if (tid < EPB && ebase + tid < E)
        y[ebase + tid] = spartial[tid] + b2[0];
}

extern "C" void kernel_launch(void* const* d_in, const int* in_sizes, int n_in,
                              void* d_out, int out_size, void* d_ws, size_t ws_size,
                              hipStream_t stream) {
    const float*     x  = (const float*)d_in[0];
    const long long* ei = (const long long*)d_in[1];
    // d_in[2] edge_attr3, d_in[3] edge_attr4, d_in[4] batch: unused by reference math
    const float* W1 = (const float*)d_in[5];
    const float* b1 = (const float*)d_in[6];
    const float* W2 = (const float*)d_in[7];
    const float* b2 = (const float*)d_in[8];
    float* y = (float*)d_out;

    const int E  = in_sizes[1] / 2;        // edge_index3 is [2, E]
    const int nw = NOUT * KDIM;            // 294912 W1 elements
    unsigned short* w1b = (unsigned short*)d_ws;   // 576 KB of scratch

    w1_to_bf16_kernel<<<(nw + 255) / 256, 256, 0, stream>>>(W1, w1b, nw);

    const int blocks = (E + EPB - 1) / EPB;  // 6250 for E=400000
    edge_mlp_wmma_kernel<<<blocks, 256, LDS_BYTES, stream>>>(
        x, ei, w1b, b1, W2, b2, y, E);
}
